// MultiHeadSelfAttention_73993696576156
// MI455X (gfx1250) — matrix operand order repaired and checked by reading, not compiled, not run on hardware
//
#include <hip/hip_runtime.h>
#include <hip/hip_bf16.h>

// ---------------------------------------------------------------------------
// Multi-head self attention, bf16 WMMA (gfx1250), flash-attention softmax,
// async global->LDS DMA staging (ASYNCcnt) for all tiles.
// B=2, S=2048, D=1024, H=16, Dh=64.
// ---------------------------------------------------------------------------

typedef __attribute__((ext_vector_type(4)))  float  f32x4;
typedef __attribute__((ext_vector_type(8)))  float  f32x8;
typedef __attribute__((ext_vector_type(4)))  __bf16 bf16x4;
typedef __attribute__((ext_vector_type(16))) __bf16 bf16x16;

union FragAB { bf16x16 v; f32x4 q[2]; };   // 32 bytes: one WMMA A or B operand per lane

#define EMBED   1024
#define NHEADS  16
#define HDIM    64
#define BATCH   2
#define SEQ     2048
#define MROWS   (BATCH * SEQ)      // 4096

// Padded LDS row stride (elements): 72 elems = 144 B = 36 banks; 16-lane
// fragment reads land on 16 distinct banks.
#define TSTR 72
#define KDEPTH 64                  // GEMM k-tile depth (2 WMMA k-steps)
#define TILE_B (128 * TSTR * 2)    // bytes per GEMM LDS tile buffer

static __device__ __forceinline__ f32x8 wmma_bf16(const FragAB& a, const FragAB& b, f32x8 c) {
  return __builtin_amdgcn_wmma_f32_16x16x32_bf16(false, a.v, false, b.v, (short)0, c, false, false);
}

// ---- CDNA5 async global->LDS DMA helpers (tracked with ASYNCcnt) -----------
static __device__ __forceinline__ uint32_t lds_off_u32(const void* p) {
  // low 32 bits of a generic LDS address are the workgroup-relative LDS offset
  return (uint32_t)(uintptr_t)p;
}
static __device__ __forceinline__ void async_b128(uint32_t lds, const __bf16* g) {
  asm volatile("global_load_async_to_lds_b128 %0, %1, off" :: "v"(lds), "v"(g) : "memory");
}
static __device__ __forceinline__ void wait_async0() {
#if defined(__has_builtin)
#if __has_builtin(__builtin_amdgcn_s_wait_asynccnt)
  __builtin_amdgcn_s_wait_asynccnt(0);
  return;
#endif
#endif
  asm volatile("s_wait_asynccnt 0x0" ::: "memory");
}

// ---------------------------------------------------------------------------
// fp32 -> bf16 conversion (4 elems / thread)
// ---------------------------------------------------------------------------
__global__ __launch_bounds__(256) void cvt_f32_to_bf16(const float* __restrict__ src,
                                                       __bf16* __restrict__ dst, int n) {
  int i = (blockIdx.x * 256 + threadIdx.x) * 4;
  if (i >= n) return;
  f32x4 v = *(const f32x4*)(src + i);
  bf16x4 o;
  o.x = (__bf16)v.x; o.y = (__bf16)v.y; o.z = (__bf16)v.z; o.w = (__bf16)v.w;
  *(bf16x4*)(dst + i) = o;
}

// ---------------------------------------------------------------------------
// fp32 [K][N] -> bf16 transposed [N][K] (32x32 LDS-tiled). Run once per weight
// so GEMM B-tiles become contiguous-row async DMA copies.
// ---------------------------------------------------------------------------
__global__ __launch_bounds__(256) void cvt_transpose_bf16(const float* __restrict__ src,
                                                          __bf16* __restrict__ dst) {
  __shared__ float tile[32][33];
  const int bx = blockIdx.x * 32;              // n base
  const int by = blockIdx.y * 32;              // k base
  const int tx = threadIdx.x & 31, ty = threadIdx.x >> 5;   // 32 x 8
#pragma unroll
  for (int i = 0; i < 32; i += 8)
    tile[ty + i][tx] = src[(size_t)(by + ty + i) * EMBED + bx + tx];
  __syncthreads();
#pragma unroll
  for (int i = 0; i < 32; i += 8)
    dst[(size_t)(bx + ty + i) * EMBED + by + tx] = (__bf16)tile[tx][ty + i];
}

// ---------------------------------------------------------------------------
// Shared 128x128 GEMM mainloop: C[128x128] += A[128xK] * Bt[128xK]^T
// A row-major [M][K]; Bt pre-transposed [N][K]. 8 waves (4Mx2N), each wave a
// 32x64 tile = 2x4 WMMA accumulators. k-tiles of depth 64 (16 WMMAs/barrier)
// staged by async DMA, double buffered. #pragma unroll 1 keeps one loop body
// so accumulators stay pinned in their WMMA register blocks (no v_mov storms).
// ---------------------------------------------------------------------------
__device__ __forceinline__ void gemm_mainloop(const __bf16* __restrict__ Ag,
                                              const __bf16* __restrict__ Bt,
                                              int Kdim, int m0, int n0,
                                              __bf16* Alds, __bf16* Blds,
                                              f32x8 acc[2][4]) {
  const int tid = threadIdx.x;
  const int wv = tid >> 5, lane = tid & 31;
  const int wm = wv & 3,   wn = wv >> 2;
  const int hf = lane >> 4, lm = lane & 15;

  // tile loader: 2 threads per row, 32 elements (4 x b128 DMA) each
  const int lrow = tid >> 1, lseg = (tid & 1) * 32;
  const __bf16* agp = Ag + (size_t)(m0 + lrow) * Kdim + lseg;
  const __bf16* bgp = Bt + (size_t)(n0 + lrow) * Kdim + lseg;
  const uint32_t aoff0 = lds_off_u32(Alds + lrow * TSTR + lseg);
  const uint32_t boff0 = lds_off_u32(Blds + lrow * TSTR + lseg);

  // prologue: DMA k-tile 0 into buffer 0
#pragma unroll
  for (int c = 0; c < 4; ++c) {
    async_b128(aoff0 + c * 16, agp + c * 8);
    async_b128(boff0 + c * 16, bgp + c * 8);
  }

  const int NK = Kdim / KDEPTH;
#pragma unroll 1
  for (int kt = 0; kt < NK; ++kt) {
    wait_async0();        // this wave's DMA for buffer (kt&1) has landed in LDS
    __syncthreads();      // publish to all waves
    if (kt + 1 < NK) {    // DMA next k-tile into the other buffer
      const __bf16* ga = agp + (kt + 1) * KDEPTH;
      const __bf16* gb = bgp + (kt + 1) * KDEPTH;
      const uint32_t la = aoff0 + ((kt + 1) & 1) * TILE_B;
      const uint32_t lb = boff0 + ((kt + 1) & 1) * TILE_B;
#pragma unroll
      for (int c = 0; c < 4; ++c) {
        async_b128(la + c * 16, ga + c * 8);
        async_b128(lb + c * 16, gb + c * 8);
      }
    }
    const __bf16* Ab = Alds + (kt & 1) * (128 * TSTR);
    const __bf16* Bb = Blds + (kt & 1) * (128 * TSTR);

#pragma unroll
    for (int ks = 0; ks < 2; ++ks) {       // two 16x16x32 k-steps per tile
      FragAB a0, a1, bfr[4];
      {
        const __bf16* p0 = Ab + (wm * 32 + lm) * TSTR + ks * 32 + 16 * hf;
        a0.q[0] = *(const f32x4*)p0; a0.q[1] = *(const f32x4*)(p0 + 8);
        const __bf16* p1 = p0 + 16 * TSTR;
        a1.q[0] = *(const f32x4*)p1; a1.q[1] = *(const f32x4*)(p1 + 8);
      }
#pragma unroll
      for (int nf = 0; nf < 4; ++nf) {
        const __bf16* p = Bb + (wn * 64 + nf * 16 + lm) * TSTR + ks * 32 + 16 * hf;
        bfr[nf].q[0] = *(const f32x4*)p; bfr[nf].q[1] = *(const f32x4*)(p + 8);
      }
#pragma unroll
      for (int nf = 0; nf < 4; ++nf) {     // 8 back-to-back WMMAs per k-step
        acc[0][nf] = wmma_bf16(a0, bfr[nf], acc[0][nf]);
        acc[1][nf] = wmma_bf16(a1, bfr[nf], acc[1][nf]);
      }
    }
  }
}

// ---------------------------------------------------------------------------
// QKV projection: z = 0/1/2 selects Wq/Wk/Wv (all pre-transposed [N][K]).
//   Q,K: [B,H,S,Dh] bf16;  V stored transposed: [B,H,Dh,S] bf16.
// ---------------------------------------------------------------------------
__global__ __launch_bounds__(256) void qkv_gemm(const __bf16* __restrict__ X,
                                                const __bf16* __restrict__ W0,
                                                const __bf16* __restrict__ W1,
                                                const __bf16* __restrict__ W2,
                                                const float* __restrict__ b0,
                                                const float* __restrict__ b1,
                                                const float* __restrict__ b2,
                                                __bf16* __restrict__ Qo,
                                                __bf16* __restrict__ Ko,
                                                __bf16* __restrict__ Vto) {
  __shared__ __align__(16) __bf16 Alds[2 * 128 * TSTR];
  __shared__ __align__(16) __bf16 Blds[2 * 128 * TSTR];

  const int which = blockIdx.z;
  const __bf16* W    = (which == 0) ? W0 : (which == 1) ? W1 : W2;
  const float*  bias = (which == 0) ? b0 : (which == 1) ? b1 : b2;

  const int m0 = blockIdx.y * 128, n0 = blockIdx.x * 128;
  f32x8 acc[2][4] = {};
  gemm_mainloop(X, W, EMBED, m0, n0, Alds, Blds, acc);

  const int tid = threadIdx.x, wv = tid >> 5, lane = tid & 31;
  const int wm = wv & 3, wn = wv >> 2, hf = lane >> 4, lm = lane & 15;

  float bvv[4];
#pragma unroll
  for (int nf = 0; nf < 4; ++nf) bvv[nf] = bias[n0 + wn * 64 + nf * 16 + lm];

#pragma unroll
  for (int mi = 0; mi < 2; ++mi)
#pragma unroll
    for (int nf = 0; nf < 4; ++nf)
#pragma unroll
      for (int v = 0; v < 8; ++v) {
        const int m = m0 + wm * 32 + mi * 16 + v + 8 * hf;  // 0..4095
        const int n = n0 + wn * 64 + nf * 16 + lm;          // 0..1023
        const float val = acc[mi][nf][v] + bvv[nf];
        const int bIdx = m >> 11, s = m & (SEQ - 1);
        const int hh = n >> 6, d = n & (HDIM - 1);
        const size_t bh = (size_t)bIdx * NHEADS + hh;
        const __bf16 o = (__bf16)val;
        if (which == 0)      Qo[(bh * SEQ + s) * HDIM + d] = o;
        else if (which == 1) Ko[(bh * SEQ + s) * HDIM + d] = o;
        else                 Vto[(bh * HDIM + d) * SEQ + s] = o;
      }
}

// ---------------------------------------------------------------------------
// Flash attention: grid (S/256, H, B). 8 waves x 32 query rows = 256 q/WG.
// Key blocks of 64, double-buffered async DMA; online softmax in registers;
// P staged to per-wave LDS scratch laid out for direct A-fragment reads.
// ---------------------------------------------------------------------------
__global__ __launch_bounds__(256) void attn_kernel(const __bf16* __restrict__ Q,
                                                   const __bf16* __restrict__ Km,
                                                   const __bf16* __restrict__ Vt,
                                                   __bf16* __restrict__ Ccat) {
  __shared__ __align__(16) __bf16 Kblk[2][64 * TSTR];     // [t][d]
  __shared__ __align__(16) __bf16 Vblk[2][64 * TSTR];     // [d][t]
  __shared__ __align__(16) __bf16 Plds[8 * 32 * TSTR];    // per-wave P scratch

  const int tid = threadIdx.x, wv = tid >> 5, lane = tid & 31;
  const int hf = lane >> 4, lm = lane & 15;
  const int h = blockIdx.y, bIdx = blockIdx.z;
  const size_t bh = (size_t)bIdx * NHEADS + h;
  const int qBase = blockIdx.x * 256 + wv * 32;

  const __bf16* Qbh = Q  + bh * SEQ * HDIM;
  const __bf16* Kbh = Km + bh * SEQ * HDIM;
  const __bf16* Vbh = Vt + bh * HDIM * SEQ;
  __bf16* Pw = Plds + wv * 32 * TSTR;

  // DMA loader: 4 threads per row, 16 elements (2 x b128) each
  const int krow = tid >> 2, kseg = (tid & 3) * 16;
  const __bf16* gK0 = Kbh + (size_t)krow * HDIM + kseg;        // + kb*64*HDIM
  const __bf16* gV0 = Vbh + (size_t)krow * SEQ + kseg;         // + kb*64
  const uint32_t lK0 = lds_off_u32(&Kblk[0][krow * TSTR + kseg]);
  const uint32_t lV0 = lds_off_u32(&Vblk[0][krow * TSTR + kseg]);
  const uint32_t kbufB = (uint32_t)(64 * TSTR * 2);            // bytes per buffer

  // Preload Q fragments (reused across all key blocks): a_q[mi][kstep]
  FragAB a_q[2][2];
#pragma unroll
  for (int mi = 0; mi < 2; ++mi) {
    const __bf16* qrow = Qbh + (size_t)(qBase + mi * 16 + lm) * HDIM;
#pragma unroll
    for (int ks = 0; ks < 2; ++ks) {
      const __bf16* p = qrow + ks * 32 + 16 * hf;
      a_q[mi][ks].q[0] = *(const f32x4*)p;
      a_q[mi][ks].q[1] = *(const f32x4*)(p + 8);
    }
  }

  f32x8 acc_o[2][4] = {};
  float mstate[2][8], lstate[2][8];
#pragma unroll
  for (int mi = 0; mi < 2; ++mi)
#pragma unroll
    for (int v = 0; v < 8; ++v) { mstate[mi][v] = -1e30f; lstate[mi][v] = 0.0f; }

  // prologue: DMA key-block 0 into buffer 0
  async_b128(lK0, gK0);      async_b128(lK0 + 16, gK0 + 8);
  async_b128(lV0, gV0);      async_b128(lV0 + 16, gV0 + 8);

  const int NKB = SEQ / 64;
#pragma unroll 1
  for (int kb = 0; kb < NKB; ++kb) {
    wait_async0();
    __syncthreads();
    if (kb + 1 < NKB) {      // DMA next key block into the other buffer
      const __bf16* gk = gK0 + (size_t)(kb + 1) * 64 * HDIM;
      const __bf16* gv = gV0 + (size_t)(kb + 1) * 64;
      const uint32_t lk = lK0 + ((kb + 1) & 1) * kbufB;
      const uint32_t lv = lV0 + ((kb + 1) & 1) * kbufB;
      async_b128(lk, gk); async_b128(lk + 16, gk + 8);
      async_b128(lv, gv); async_b128(lv + 16, gv + 8);
    }
    const __bf16* Kb = Kblk[kb & 1];
    const __bf16* Vb = Vblk[kb & 1];

    // ---- scores = Q @ K^T (K block [t][d] is exactly the B-operand layout)
    f32x8 s[2][4] = {};
#pragma unroll
    for (int ks = 0; ks < 2; ++ks) {
      FragAB bk[4];
#pragma unroll
      for (int nf = 0; nf < 4; ++nf) {
        const __bf16* p = Kb + (nf * 16 + lm) * TSTR + ks * 32 + 16 * hf;
        bk[nf].q[0] = *(const f32x4*)p; bk[nf].q[1] = *(const f32x4*)(p + 8);
      }
#pragma unroll
      for (int nf = 0; nf < 4; ++nf) {
        s[0][nf] = wmma_bf16(a_q[0][ks], bk[nf], s[0][nf]);
        s[1][nf] = wmma_bf16(a_q[1][ks], bk[nf], s[1][nf]);
      }
    }
#pragma unroll
    for (int mi = 0; mi < 2; ++mi)
#pragma unroll
      for (int nf = 0; nf < 4; ++nf) s[mi][nf] = s[mi][nf] * 0.125f;  // 1/sqrt(Dh)

    // ---- online softmax (rows live in 16-lane halves; reduce with shfl_xor)
#pragma unroll
    for (int mi = 0; mi < 2; ++mi) {
#pragma unroll
      for (int v = 0; v < 8; ++v) {
        float mx = fmaxf(fmaxf(s[mi][0][v], s[mi][1][v]), fmaxf(s[mi][2][v], s[mi][3][v]));
        mx = fmaxf(mx, __shfl_xor(mx, 1));
        mx = fmaxf(mx, __shfl_xor(mx, 2));
        mx = fmaxf(mx, __shfl_xor(mx, 4));
        mx = fmaxf(mx, __shfl_xor(mx, 8));
        const float mold = mstate[mi][v];
        const float mnew = fmaxf(mold, mx);
        mstate[mi][v] = mnew;
        const float sc = __expf(mold - mnew);
        float ps = 0.0f;
        const int prow = (mi * 16 + v + 8 * hf) * TSTR + lm;
#pragma unroll
        for (int nf = 0; nf < 4; ++nf) {
          const float pv = __expf(s[mi][nf][v] - mnew);
          ps += pv;
          Pw[prow + nf * 16] = (__bf16)pv;
          acc_o[mi][nf][v] *= sc;
        }
        ps += __shfl_xor(ps, 1);
        ps += __shfl_xor(ps, 2);
        ps += __shfl_xor(ps, 4);
        ps += __shfl_xor(ps, 8);
        lstate[mi][v] = lstate[mi][v] * sc + ps;
      }
    }

    // ---- acc_o += P @ V (same-wave LDS ordering covers Pw; no barrier)
#pragma unroll
    for (int t0 = 0; t0 < 64; t0 += 32) {
      FragAB ap[2], bv[4];
#pragma unroll
      for (int mi = 0; mi < 2; ++mi) {
        const __bf16* p = Pw + (mi * 16 + lm) * TSTR + t0 + 16 * hf;
        ap[mi].q[0] = *(const f32x4*)p; ap[mi].q[1] = *(const f32x4*)(p + 8);
      }
#pragma unroll
      for (int nf = 0; nf < 4; ++nf) {
        const __bf16* p = Vb + (nf * 16 + lm) * TSTR + t0 + 16 * hf;
        bv[nf].q[0] = *(const f32x4*)p; bv[nf].q[1] = *(const f32x4*)(p + 8);
      }
#pragma unroll
      for (int nf = 0; nf < 4; ++nf) {
        acc_o[0][nf] = wmma_bf16(ap[0], bv[nf], acc_o[0][nf]);
        acc_o[1][nf] = wmma_bf16(ap[1], bv[nf], acc_o[1][nf]);
      }
    }
  }

  // ---- normalize and store concat layout [B, S, H*Dh] bf16
#pragma unroll
  for (int mi = 0; mi < 2; ++mi)
#pragma unroll
    for (int v = 0; v < 8; ++v) {
      const float inv = 1.0f / lstate[mi][v];
      const int qrow = qBase + mi * 16 + v + 8 * hf;
      const size_t base = ((size_t)(bIdx * SEQ + qrow)) * EMBED + h * HDIM;
#pragma unroll
      for (int nf = 0; nf < 4; ++nf)
        Ccat[base + nf * 16 + lm] = (__bf16)(acc_o[mi][nf][v] * inv);
    }
}

// ---------------------------------------------------------------------------
// Output projection: out = concat @ Wo^T(pre-transposed) + bo (fp32 output)
// ---------------------------------------------------------------------------
__global__ __launch_bounds__(256) void proj_gemm(const __bf16* __restrict__ A,
                                                 const __bf16* __restrict__ Wt,
                                                 const float* __restrict__ bias,
                                                 float* __restrict__ Out) {
  __shared__ __align__(16) __bf16 Alds[2 * 128 * TSTR];
  __shared__ __align__(16) __bf16 Blds[2 * 128 * TSTR];

  const int m0 = blockIdx.y * 128, n0 = blockIdx.x * 128;
  f32x8 acc[2][4] = {};
  gemm_mainloop(A, Wt, EMBED, m0, n0, Alds, Blds, acc);

  const int tid = threadIdx.x, wv = tid >> 5, lane = tid & 31;
  const int wm = wv & 3, wn = wv >> 2, hf = lane >> 4, lm = lane & 15;

  float bvv[4];
#pragma unroll
  for (int nf = 0; nf < 4; ++nf) bvv[nf] = bias[n0 + wn * 64 + nf * 16 + lm];

#pragma unroll
  for (int mi = 0; mi < 2; ++mi)
#pragma unroll
    for (int nf = 0; nf < 4; ++nf)
#pragma unroll
      for (int v = 0; v < 8; ++v) {
        const int m = m0 + wm * 32 + mi * 16 + v + 8 * hf;
        const int n = n0 + wn * 64 + nf * 16 + lm;
        Out[(size_t)m * EMBED + n] = acc[mi][nf][v] + bvv[nf];
      }
}

// ---------------------------------------------------------------------------
extern "C" void kernel_launch(void* const* d_in, const int* in_sizes, int n_in,
                              void* d_out, int out_size, void* d_ws, size_t ws_size,
                              hipStream_t stream) {
  const float* X   = (const float*)d_in[0];
  const float* Wq  = (const float*)d_in[1];
  const float* bq  = (const float*)d_in[2];
  const float* Wk  = (const float*)d_in[3];
  const float* bk  = (const float*)d_in[4];
  const float* Wv  = (const float*)d_in[5];
  const float* bvp = (const float*)d_in[6];
  const float* Wo  = (const float*)d_in[7];
  const float* bo  = (const float*)d_in[8];

  const int NX = MROWS * EMBED;    // 4,194,304
  const int NW = EMBED * EMBED;    // 1,048,576

  char* ws = (char*)d_ws;
  __bf16* Xbf = (__bf16*)(ws);                       //  8 MiB
  __bf16* Wqb = (__bf16*)(ws + (size_t)8  * 1024 * 1024);  // transposed [N][K]
  __bf16* Wkb = Wqb + NW;
  __bf16* Wvb = Wkb + NW;
  __bf16* Wob = Wvb + NW;
  __bf16* Qb  = (__bf16*)(ws + (size_t)16 * 1024 * 1024);
  __bf16* Kb  = Qb + NX;
  __bf16* Vtb = Kb + NX;
  __bf16* Cc  = Vtb + NX;                            // ends at 48 MiB

  cvt_f32_to_bf16<<<NX / 4 / 256, 256, 0, stream>>>(X, Xbf, NX);
  cvt_transpose_bf16<<<dim3(32, 32), 256, 0, stream>>>(Wq, Wqb);
  cvt_transpose_bf16<<<dim3(32, 32), 256, 0, stream>>>(Wk, Wkb);
  cvt_transpose_bf16<<<dim3(32, 32), 256, 0, stream>>>(Wv, Wvb);
  cvt_transpose_bf16<<<dim3(32, 32), 256, 0, stream>>>(Wo, Wob);

  qkv_gemm<<<dim3(EMBED / 128, MROWS / 128, 3), 256, 0, stream>>>(
      Xbf, Wqb, Wkb, Wvb, bq, bk, bvp, Qb, Kb, Vtb);

  attn_kernel<<<dim3(SEQ / 256, NHEADS, BATCH), 256, 0, stream>>>(Qb, Kb, Vtb, Cc);

  proj_gemm<<<dim3(EMBED / 128, MROWS / 128), 256, 0, stream>>>(
      Cc, Wob, bo, (float*)d_out);
}
